// NeuralCDE_24893630447785
// MI455X (gfx1250) — matrix operand order repaired and checked by reading, not compiled, not run on hardware
//
#include <hip/hip_runtime.h>
#include <hip/hip_bf16.h>

// ---------------------------------------------------------------------------
// Neural CDE on MI455X (gfx1250): persistent cooperative kernel.
//   - bf16 WMMA (v_wmma_f32_16x16x32_bf16) matvecs, f32 accumulate
//   - weights live in L2 (36 MB bf16, swizzled tiles in d_ws)
//   - sense-reversing global barrier with s_sleep backoff
//   - stage 3 processes two row-tiles per wave sharing one B load (dual accum)
// ---------------------------------------------------------------------------

typedef __attribute__((ext_vector_type(16))) __bf16 v16bf;
typedef __attribute__((ext_vector_type(8)))  float  v8f;

#define NWG    64
#define NTHR   256
#define GT     (NWG * NTHR)
#define TSTEPS 4096
#define DDIM   64
#define HDIM   256
#define WDIM   1024

#define FW0_TILES (64 * 8)      // 1024x256  -> 64 row-tiles x 8 K-chunks
#define FW1_TILES (64 * 32)     // 1024x1024
#define FW2_TILES (1024 * 32)   // 16384x1024

struct P {
  const float *ts, *cd, *ccf, *cb, *ca;
  const float *iw0, *ib0, *iw1, *ib1, *iw2, *ib2;
  const float *fw0, *fb0, *fw1, *fb1, *fw2, *fb2, *lw, *lb;
  float    *out;
  unsigned *bar;                 // [0]=arrive, [1]=epoch
  float    *kv, *h1g, *h2g, *y0g;
  __bf16   *fw0s, *fw1s, *fw2s;  // swizzled bf16 tiles
};

__device__ __forceinline__ float softplusf(float x) {
  return fmaxf(x, 0.f) + log1pf(expf(-fabsf(x)));
}

// grid-wide sense/epoch barrier (all NWG blocks resident)
__device__ __forceinline__ void gbar(unsigned* bar, unsigned* ep_local) {
  __syncthreads();
  if (threadIdx.x == 0) {
    unsigned e = *ep_local;
    unsigned prev = __hip_atomic_fetch_add(&bar[0], 1u, __ATOMIC_ACQ_REL,
                                           __HIP_MEMORY_SCOPE_AGENT);
    if (prev == NWG - 1u) {
      __hip_atomic_store(&bar[0], 0u, __ATOMIC_RELAXED, __HIP_MEMORY_SCOPE_AGENT);
      __hip_atomic_fetch_add(&bar[1], 1u, __ATOMIC_RELEASE, __HIP_MEMORY_SCOPE_AGENT);
    } else {
      while (__hip_atomic_load(&bar[1], __ATOMIC_ACQUIRE,
                               __HIP_MEMORY_SCOPE_AGENT) <= e) {
        __builtin_amdgcn_s_sleep(2);
      }
    }
    *ep_local = e + 1;
  }
  __syncthreads();
}

// One 16-row output tile of W @ x over K = KC*32, bf16 WMMA, f32 accum.
// Wsw: swizzled tiles (1KB each: lane*32B contiguous). xswz: LDS, replicated-B
// form: per 32-K chunk, [h=0]=K0..15, [h=1]=K16..31, 16 bf16 each.
template <int KC>
__device__ __forceinline__ v8f mv_tile(const __bf16* __restrict__ Wsw,
                                       size_t tileBase, const __bf16* xswz,
                                       int lane) {
  v8f acc = {0.f, 0.f, 0.f, 0.f, 0.f, 0.f, 0.f, 0.f};
  const int h = lane >> 4;
  const __bf16* wp = Wsw + tileBase + (size_t)lane * 16;
#pragma unroll 8
  for (int c = 0; c < KC; ++c) {
    v16bf a = *reinterpret_cast<const v16bf*>(wp + (size_t)c * 512);
    v16bf b = *reinterpret_cast<const v16bf*>(xswz + (c * 2 + h) * 16);
    acc = __builtin_amdgcn_wmma_f32_16x16x32_bf16(false, a, false, b,
                                                  (short)0, acc, false, false);
  }
  return acc;
}

// Two consecutive row-tiles sharing each B load; two independent accum chains.
template <int KC>
__device__ __forceinline__ void mv_tile2(const __bf16* __restrict__ Wsw,
                                         size_t base0, const __bf16* xswz,
                                         int lane, v8f& acc0, v8f& acc1) {
  const int h = lane >> 4;
  const __bf16* wp0 = Wsw + base0 + (size_t)lane * 16;
  const __bf16* wp1 = wp0 + (size_t)KC * 512;  // next tile row's chunk block
#pragma unroll 4
  for (int c = 0; c < KC; ++c) {
    v16bf b  = *reinterpret_cast<const v16bf*>(xswz + (c * 2 + h) * 16);
    v16bf a0 = *reinterpret_cast<const v16bf*>(wp0 + (size_t)c * 512);
    v16bf a1 = *reinterpret_cast<const v16bf*>(wp1 + (size_t)c * 512);
    acc0 = __builtin_amdgcn_wmma_f32_16x16x32_bf16(false, a0, false, b,
                                                   (short)0, acc0, false, false);
    acc1 = __builtin_amdgcn_wmma_f32_16x16x32_bf16(false, a1, false, b,
                                                   (short)0, acc1, false, false);
  }
}

__global__ __launch_bounds__(NTHR, 1) void cde_persistent(P p) {
  __shared__ __align__(32) __bf16 xb[8 * 2 * 16];     // y-input, K=256
  __shared__ __align__(32) __bf16 hswz[32 * 2 * 16];  // hidden, K=1024
  __shared__ float y_lds[HDIM];
  __shared__ float kacc[HDIM];
  __shared__ float cder[DDIM];
  __shared__ float part[16][2];
  __shared__ float red[NTHR];

  const int tid  = threadIdx.x;
  const int lane = tid & 31;
  const int wave = tid >> 5;
  const int blk  = blockIdx.x;
  const int gtid = blk * NTHR + tid;
  unsigned ep = 0;

  // ---- phase A: convert f32 weights -> bf16 WMMA-swizzled tiles in ws ----
  {
    const int tot = (FW0_TILES + FW1_TILES + FW2_TILES) * 32;
    for (int task = gtid; task < tot; task += GT) {
      int tile = task >> 5, ln = task & 31;
      const float* src; __bf16* dst; int K, Kc, lt;
      if (tile < FW0_TILES) {
        src = p.fw0; dst = p.fw0s; K = HDIM; Kc = 8;  lt = tile;
      } else if (tile < FW0_TILES + FW1_TILES) {
        src = p.fw1; dst = p.fw1s; K = WDIM; Kc = 32; lt = tile - FW0_TILES;
      } else {
        src = p.fw2; dst = p.fw2s; K = WDIM; Kc = 32;
        lt = tile - FW0_TILES - FW1_TILES;
      }
      int rt = lt / Kc, kc = lt % Kc;
      int row = rt * 16 + (ln & 15);
      int h   = ln >> 4;
      v16bf v;
#pragma unroll
      for (int s2 = 0; s2 < 16; ++s2) {
        // contiguous A k order used here (B matches): lane<16 K {0..15}, lane>=16 {16..31}
        int k32 = s2 + (h << 4);
        v[s2] = (__bf16)src[(size_t)row * K + kc * 32 + k32];
      }
      *reinterpret_cast<v16bf*>(dst + (size_t)lt * 512 + ln * 16) = v;
    }
  }
  gbar(p.bar, &ep);

  // ---- phase B: initial MLP (relu, f32 scalar; one-time) ----
  for (int o = gtid; o < WDIM; o += GT) {
    float a = p.ib0[o];
    for (int k = 0; k < DDIM; ++k) a = fmaf(p.iw0[(size_t)o * DDIM + k], p.ca[k], a);
    p.h1g[o] = fmaxf(a, 0.f);
  }
  gbar(p.bar, &ep);
  for (int o = gtid; o < WDIM; o += GT) {
    float a = p.ib1[o];
    for (int k = 0; k < WDIM; ++k) a = fmaf(p.iw1[(size_t)o * WDIM + k], p.h1g[k], a);
    p.h2g[o] = fmaxf(a, 0.f);
  }
  gbar(p.bar, &ep);
  for (int o = gtid; o < HDIM; o += GT) {
    float a = p.ib2[o];
    for (int k = 0; k < WDIM; ++k) a = fmaf(p.iw2[(size_t)o * WDIM + k], p.h2g[k], a);
    p.y0g[o] = a;
  }
  gbar(p.bar, &ep);

  // local state: y, kacc, first B-swizzled input
  {
    float y = p.y0g[tid];
    y_lds[tid] = y;
    kacc[tid]  = 0.f;
    int c = tid >> 5, k32 = tid & 31;
    xb[(c * 2 + (k32 >> 4)) * 16 + (k32 & 15)] = (__bf16)y;
  }
  const float dt = p.ts[1] - p.ts[0];

#pragma unroll 1
  for (int s = 0; s < TSTEPS; ++s) {
#pragma unroll 1
    for (int e = 0; e < 3; ++e) {
      __syncthreads();  // xb ready
      // ---- stage 1: h1 = softplus(fw0 @ x + fb0), 64 tiles -> 1 per block ----
      if (wave == 0) {
        v8f c1 = mv_tile<8>(p.fw0s, (size_t)blk * 8 * 512, xb, lane);
        if ((lane & 15) == 0) {
          int rb = (lane >> 4) * 8;
#pragma unroll
          for (int r = 0; r < 8; ++r) {
            int o = blk * 16 + rb + r;
            p.h1g[o] = softplusf(c1[r] + p.fb0[o]);
          }
        }
      }
      gbar(p.bar, &ep);
      // ---- stage 2: h2 = softplus(fw1 @ h1 + fb1) ----
      for (int k = tid; k < WDIM; k += NTHR) {
        int c = k >> 5, k32 = k & 31;
        hswz[(c * 2 + (k32 >> 4)) * 16 + (k32 & 15)] = (__bf16)p.h1g[k];
      }
      __syncthreads();
      if (wave == 0) {
        v8f c2 = mv_tile<32>(p.fw1s, (size_t)blk * 32 * 512, hswz, lane);
        if ((lane & 15) == 0) {
          int rb = (lane >> 4) * 8;
#pragma unroll
          for (int r = 0; r < 8; ++r) {
            int o = blk * 16 + rb + r;
            p.h2g[o] = softplusf(c2[r] + p.fb1[o]);
          }
        }
      }
      gbar(p.bar, &ep);
      // ---- stage 3: k = tanh(fw2 @ h2 + fb2).reshape(H,D) @ cderiv * dt ----
      for (int k = tid; k < WDIM; k += NTHR) {
        int c = k >> 5, k32 = k & 31;
        hswz[(c * 2 + (k32 >> 4)) * 16 + (k32 & 15)] = (__bf16)p.h2g[k];
      }
      if (tid < DDIM) {
        int i = (s < TSTEPS - 2) ? s : (TSTEPS - 2);
        float frac = (e == 0) ? 0.f : (e == 1) ? 0.5f : 0.75f;
        float f = ((float)(s - i) + frac) * dt;
        size_t off = (size_t)i * DDIM + tid;
        cder[tid] = p.cb[off] + f * (2.f * p.ccf[off] + 3.f * f * p.cd[off]);
      }
      __syncthreads();
      {
        int tl = wave * 2;          // two tiles per wave, shared B loads
        int r  = blk * 16 + tl;     // global 16-row tile of the 16384 outputs
        v8f c30 = {0.f, 0.f, 0.f, 0.f, 0.f, 0.f, 0.f, 0.f};
        v8f c31 = {0.f, 0.f, 0.f, 0.f, 0.f, 0.f, 0.f, 0.f};
        mv_tile2<32>(p.fw2s, (size_t)r * 32 * 512, hswz, lane, c30, c31);
        if ((lane & 15) == 0) {
          int half = lane >> 4;
#pragma unroll 1
          for (int u = 0; u < 2; ++u) {
            int mb = (r + u) * 16 + half * 8;
            float pp = 0.f;
#pragma unroll
            for (int rr = 0; rr < 8; ++rr) {
              int m = mb + rr;
              float cv = (u == 0) ? c30[rr] : c31[rr];
              pp += tanhf(cv + p.fb2[m]) * cder[m & (DDIM - 1)];
            }
            part[tl + u][half] = pp;
          }
        }
      }
      __syncthreads();
      if (tid < 4) {  // block owns k-rows blk*4 .. blk*4+3 exclusively
        float acc = 0.f;
#pragma unroll
        for (int q = 0; q < 4; ++q) acc += part[tid * 4 + q][0] + part[tid * 4 + q][1];
        p.kv[blk * 4 + tid] = acc * dt;  // vf_prod's dt0 factor
      }
      gbar(p.bar, &ep);
      // ---- post: RK3 state update (replicated identically in every block) ----
      {
        float kvv = p.kv[tid];
        float xn;
        if (e == 0) {
          kacc[tid] = (2.f / 9.f) * kvv;
          xn = y_lds[tid] + 0.5f * kvv;
        } else if (e == 1) {
          kacc[tid] += (1.f / 3.f) * kvv;
          xn = y_lds[tid] + 0.75f * kvv;
        } else {
          float yn = y_lds[tid] + dt * (kacc[tid] + (4.f / 9.f) * kvv);
          y_lds[tid] = yn;
          xn = yn;
        }
        int c = tid >> 5, k32 = tid & 31;
        xb[(c * 2 + (k32 >> 4)) * 16 + (k32 & 15)] = (__bf16)xn;
      }
    }
  }

  // ---- final readout: pred = sigmoid(lw @ y + lb) ----
  if (blk == 0) {
    red[tid] = p.lw[tid] * y_lds[tid];
    __syncthreads();
    for (int w = NTHR / 2; w > 0; w >>= 1) {
      if (tid < w) red[tid] += red[tid + w];
      __syncthreads();
    }
    if (tid == 0) {
      float z = red[0] + p.lb[0];
      p.out[0] = 1.f / (1.f + expf(-z));
    }
  }
}

extern "C" void kernel_launch(void* const* d_in, const int* in_sizes, int n_in,
                              void* d_out, int out_size, void* d_ws, size_t ws_size,
                              hipStream_t stream) {
  (void)in_sizes; (void)n_in; (void)out_size; (void)ws_size;
  char* ws = (char*)d_ws;
  P p;
  p.ts  = (const float*)d_in[0];
  p.cd  = (const float*)d_in[1];
  p.ccf = (const float*)d_in[2];
  p.cb  = (const float*)d_in[3];
  p.ca  = (const float*)d_in[4];
  p.iw0 = (const float*)d_in[5];  p.ib0 = (const float*)d_in[6];
  p.iw1 = (const float*)d_in[7];  p.ib1 = (const float*)d_in[8];
  p.iw2 = (const float*)d_in[9];  p.ib2 = (const float*)d_in[10];
  p.fw0 = (const float*)d_in[11]; p.fb0 = (const float*)d_in[12];
  p.fw1 = (const float*)d_in[13]; p.fb1 = (const float*)d_in[14];
  p.fw2 = (const float*)d_in[15]; p.fb2 = (const float*)d_in[16];
  p.lw  = (const float*)d_in[17]; p.lb  = (const float*)d_in[18];
  p.out = (float*)d_out;

  p.bar = (unsigned*)(ws + 0);
  p.kv  = (float*)(ws + 256);
  p.h1g = (float*)(ws + 256 + 1024);
  p.h2g = (float*)(ws + 256 + 1024 + 4096);
  p.y0g = (float*)(ws + 256 + 1024 + 8192);
  const size_t wOff = 16384;
  p.fw0s = (__bf16*)(ws + wOff);
  p.fw1s = (__bf16*)(ws + wOff + (size_t)FW0_TILES * 1024);
  p.fw2s = (__bf16*)(ws + wOff + (size_t)(FW0_TILES + FW1_TILES) * 1024);

  hipMemsetAsync(ws, 0, 256, stream);  // reset barrier state every launch
  hipLaunchKernelGGL(cde_persistent, dim3(NWG), dim3(NTHR), 0, stream, p);
}
